// GNNBiomarkerModel_40432822124917
// MI455X (gfx1250) — compile-verified, matrix operand order checked by reading, not run on hardware
//
#include <hip/hip_runtime.h>
#include <stdint.h>

// ---------- types ----------
typedef __bf16 bf16_t;
typedef __attribute__((ext_vector_type(16))) __bf16 v16bf;
typedef __attribute__((ext_vector_type(8)))  __bf16 v8bf;
typedef __attribute__((ext_vector_type(8)))  float  v8f;

#define HID     96
#define IN_DIM  64
#define BN_EPS  1e-5f

// ---------- WMMA helpers ----------
__device__ __forceinline__ v8f wmma_bf16(v16bf a, v16bf b, v8f c) {
    // v_wmma_f32_16x16x32_bf16  (neg_a, A, neg_b, B, c_mod, C, reuse_a, reuse_b)
    return __builtin_amdgcn_wmma_f32_16x16x32_bf16(false, a, false, b, (short)0, c, false, false);
}

// 16-bit A/B fragment, 16x32 tile, row-major source with row fixed per lane.
// Lane half (lane>=16) selects +8 K offset; VGPR0-3 hold [k0+half*8 .. +7],
// VGPR4-7 hold [k0+16+half*8 .. +7]. Two contiguous 16B loads.
__device__ __forceinline__ v16bf load_frag_bf16(const bf16_t* p) {
    v8bf lo = *(const v8bf*)(p);
    v8bf hi = *(const v8bf*)(p + 16);
    v16bf r;
#pragma unroll
    for (int i = 0; i < 8; ++i) { r[i] = lo[i]; r[i + 8] = hi[i]; }
    return r;
}

__device__ __forceinline__ v16bf load_frag_f32(const float* p) {
    v16bf r;
#pragma unroll
    for (int i = 0; i < 8; ++i) { r[i] = (bf16_t)p[i]; r[i + 8] = (bf16_t)p[16 + i]; }
    return r;
}

// ---------- utility kernels ----------
__global__ void k_zero(float* p, int n) {
    int i = blockIdx.x * blockDim.x + threadIdx.x;
    if (i < n) p[i] = 0.0f;
}

// one thread per (edge, feature): coalesced gather from h[src], atomic scatter to agg[dst]
__global__ void k_scatter(const float* __restrict__ h, const int* __restrict__ src,
                          const int* __restrict__ dst, float* __restrict__ agg,
                          long long total) {
    long long i = (long long)blockIdx.x * blockDim.x + threadIdx.x;
    if (i >= total) return;
    int e = (int)(i / HID);
    int f = (int)(i % HID);
    int s = src[e], d = dst[e];
    atomicAdd(&agg[d * HID + f], h[s * HID + f]);
}

__global__ void k_deg(const int* __restrict__ dst, float* __restrict__ deg, int E) {
    int e = blockIdx.x * blockDim.x + threadIdx.x;
    if (e < E) atomicAdd(&deg[dst[e]], 1.0f);
}

// agg /= max(deg,1) and convert to bf16
__global__ void k_agg_finalize(const float* __restrict__ agg, const float* __restrict__ deg,
                               bf16_t* __restrict__ aggb, int NH) {
    int i = blockIdx.x * blockDim.x + threadIdx.x;
    if (i >= NH) return;
    int r = i / HID;
    float d = deg[r];
    d = d > 1.0f ? d : 1.0f;
    aggb[i] = (bf16_t)(agg[i] / d);
}

// ---------- embedding GEMM: h = relu(x @ W_emb + b), x:[N,64] f32 ----------
// requires n >= 16; last tile overlaps backwards (idempotent duplicate writes)
__global__ void __launch_bounds__(256)
k_embed(const float* __restrict__ x, const float* __restrict__ W,
        const float* __restrict__ b, float* __restrict__ h,
        bf16_t* __restrict__ hb, int n) {
    __shared__ __align__(16) bf16_t sW[HID * IN_DIM]; // transposed: [n][k]
    __shared__ float sb[HID];
    int t = threadIdx.x;
    for (int i = t; i < HID * IN_DIM; i += 256) {
        int k = i / HID, nn = i % HID;            // W is [IN_DIM, HID]
        sW[nn * IN_DIM + k] = (bf16_t)W[i];
    }
    for (int i = t; i < HID; i += 256) sb[i] = b[i];
    __syncthreads();

    int wave = t >> 5, lane = t & 31;
    int row0 = (blockIdx.x * 8 + wave) * 16;
    if (row0 >= n) return;
    if (row0 + 16 > n) row0 = n - 16;              // overlapping tail tile
    int half = lane >> 4, lr = lane & 15;

    const float* xr = x + (row0 + lr) * IN_DIM;
    v16bf a0 = load_frag_f32(xr + 0  + half * 8);  // K 0..31
    v16bf a1 = load_frag_f32(xr + 32 + half * 8);  // K 32..63

    int base = (row0 + half * 8) * HID + lr;       // D row stride = HID
    float*  hp  = h  + base;
    bf16_t* hbp = hb + base;

#pragma unroll
    for (int ct = 0; ct < 6; ++ct) {
        int col0 = ct * 16;
        const bf16_t* wc = &sW[(col0 + lr) * IN_DIM];
        v8f c = {};                                // inline-0 SRC2 on first WMMA
        c = wmma_bf16(a0, load_frag_bf16(wc + 0  + half * 8), c);
        c = wmma_bf16(a1, load_frag_bf16(wc + 32 + half * 8), c);
        float bias = sb[col0 + lr];                // bias folded in epilogue
#pragma unroll
        for (int v = 0; v < 8; ++v) {
            float val = fmaxf(c[v] + bias, 0.0f);
            hp[col0 + v * HID]  = val;             // immediate-offset stores
            hbp[col0 + v * HID] = (bf16_t)val;
        }
    }
}

// ---------- SAGE layer GEMM: hlin = aggb @ Wl + bl + hb @ Wr ----------
__global__ void __launch_bounds__(256)
k_gemm(const bf16_t* __restrict__ aggb, const bf16_t* __restrict__ hb,
       const float* __restrict__ Wl, const float* __restrict__ bl,
       const float* __restrict__ Wr, float* __restrict__ hlin, int n) {
    __shared__ __align__(16) bf16_t sWl[HID * HID]; // transposed [n][k]
    __shared__ __align__(16) bf16_t sWr[HID * HID];
    __shared__ float sb[HID];
    int t = threadIdx.x;
    for (int i = t; i < HID * HID; i += 256) {
        int k = i / HID, nn = i % HID;             // W is [HID, HID] (k-major)
        sWl[nn * HID + k] = (bf16_t)Wl[i];
        sWr[nn * HID + k] = (bf16_t)Wr[i];
    }
    for (int i = t; i < HID; i += 256) sb[i] = bl[i];
    __syncthreads();

    int wave = t >> 5, lane = t & 31;
    int row0 = (blockIdx.x * 8 + wave) * 16;
    if (row0 >= n) return;
    if (row0 + 16 > n) row0 = n - 16;              // overlapping tail tile
    int half = lane >> 4, lr = lane & 15;

    const bf16_t* ar = aggb + (row0 + lr) * HID;
    const bf16_t* hr = hb   + (row0 + lr) * HID;

    // Hoist A fragments (k0 = 0,32,64) for both operands; reuse across 6 col tiles.
    v16bf aA[3], aH[3];
#pragma unroll
    for (int ks = 0; ks < 3; ++ks) {
        aA[ks] = load_frag_bf16(ar + ks * 32 + half * 8);
        aH[ks] = load_frag_bf16(hr + ks * 32 + half * 8);
    }

    float* outp = hlin + (row0 + half * 8) * HID + lr;

#pragma unroll
    for (int ct = 0; ct < 6; ++ct) {
        int col0 = ct * 16;
        const bf16_t* wl = &sWl[(col0 + lr) * HID];
        const bf16_t* wr = &sWr[(col0 + lr) * HID];
        v8f c = {};                                // inline-0 SRC2 on first WMMA
#pragma unroll
        for (int ks = 0; ks < 3; ++ks) {
            c = wmma_bf16(aA[ks], load_frag_bf16(wl + ks * 32 + half * 8), c);
            c = wmma_bf16(aH[ks], load_frag_bf16(wr + ks * 32 + half * 8), c);
        }
        float bias = sb[col0 + lr];                // bias folded in epilogue
#pragma unroll
        for (int v = 0; v < 8; ++v)
            outp[col0 + v * HID] = c[v] + bias;    // immediate-offset stores
    }
}

// ---------- BatchNorm: stats (sum, sumsq per feature) then apply+relu ----------
__global__ void k_bnstats(const float* __restrict__ hlin, float* __restrict__ stats, int n) {
    int f = threadIdx.x;                           // 96 threads: coalesced row reads
    int stride = gridDim.x;
    float s = 0.0f, s2 = 0.0f;
    for (int r = blockIdx.x; r < n; r += stride) {
        float v = hlin[r * HID + f];
        int rn = r + stride;
        if (rn < n) __builtin_prefetch(&hlin[rn * HID + f], 0, 0); // global_prefetch_b8
        s += v; s2 += v * v;
    }
    atomicAdd(&stats[f], s);
    atomicAdd(&stats[HID + f], s2);
}

__global__ void k_bnapply(const float* __restrict__ hlin, const float* __restrict__ stats,
                          const float* __restrict__ gamma, const float* __restrict__ beta,
                          float* __restrict__ h, bf16_t* __restrict__ hb,
                          int NH, float invN) {
    int i = blockIdx.x * blockDim.x + threadIdx.x;
    if (i >= NH) return;
    int f = i % HID;
    float mean = stats[f] * invN;
    float var  = stats[HID + f] * invN - mean * mean;
    float v = (hlin[i] - mean) * rsqrtf(var + BN_EPS) * gamma[f] + beta[f];
    v = fmaxf(v, 0.0f);
    h[i]  = v;
    hb[i] = (bf16_t)v;
}

// ---------- head MLP on 512 central nodes ----------
__global__ void k_head(const float* __restrict__ h, const int* __restrict__ cidx,
                       const float* __restrict__ W1, const float* __restrict__ b1,
                       const float* __restrict__ W2, const float* __restrict__ b2,
                       float* __restrict__ out) {
    __shared__ float xrow[HID];
    __shared__ float red[128];
    int g = blockIdx.x, t = threadIdx.x;
    int c = cidx[g];
    if (t < HID) xrow[t] = h[c * HID + t];
    __syncthreads();
    float p = 0.0f;
    if (t < HID) {
        float s = b1[t];
        for (int k = 0; k < HID; ++k) s += xrow[k] * W1[k * HID + t];
        s = fmaxf(s, 0.0f);
        p = s * W2[t];                             // OUT_DIM == 1
    }
    red[t] = p;
    __syncthreads();
    for (int off = 64; off > 0; off >>= 1) {
        if (t < off) red[t] += red[t + off];
        __syncthreads();
    }
    if (t == 0) out[g] = red[0] + b2[0];
}

// ---------- orchestration ----------
extern "C" void kernel_launch(void* const* d_in, const int* in_sizes, int n_in,
                              void* d_out, int out_size, void* d_ws, size_t ws_size,
                              hipStream_t stream) {
    const float* x       = (const float*)d_in[0];
    const int*   edge    = (const int*)d_in[1];
    // d_in[2] = batch (unused: single graph)
    const int*   cidx    = (const int*)d_in[3];
    const float* W_emb   = (const float*)d_in[4];
    const float* b_emb   = (const float*)d_in[5];
    const float* conv_wl = (const float*)d_in[6];
    const float* conv_bl = (const float*)d_in[7];
    const float* conv_wr = (const float*)d_in[8];
    const float* gamma   = (const float*)d_in[9];
    const float* beta    = (const float*)d_in[10];
    const float* W1      = (const float*)d_in[11];
    const float* b1      = (const float*)d_in[12];
    const float* W2      = (const float*)d_in[13];
    const float* b2      = (const float*)d_in[14];
    float* out = (float*)d_out;

    const int N = in_sizes[0] / IN_DIM;
    const int E = in_sizes[1] / 2;
    const int G = in_sizes[3];
    const int* src = edge;
    const int* dst = edge + E;
    const int NH = N * HID;

    // workspace carve-out (256B aligned regions)
    uint8_t* wp = (uint8_t*)d_ws;
    auto take = [&](size_t bytes) { void* r = wp; wp += (bytes + 255) & ~(size_t)255; return r; };
    float*  h     = (float*)take((size_t)NH * 4);
    float*  hlin  = (float*)take((size_t)NH * 4);   // doubles as agg accumulator
    bf16_t* hb    = (bf16_t*)take((size_t)NH * 2);
    bf16_t* aggb  = (bf16_t*)take((size_t)NH * 2);
    float*  deg   = (float*)take((size_t)N * 4);
    float*  stats = (float*)take(2 * HID * 4);

    const int tiles       = (N + 15) / 16;
    const int gemm_blocks = (tiles + 7) / 8;       // 8 waves/block, 1 row-tile each

    k_embed<<<gemm_blocks, 256, 0, stream>>>(x, W_emb, b_emb, h, hb, N);

    for (int L = 0; L < 3; ++L) {
        k_zero<<<(NH + 255) / 256, 256, 0, stream>>>(hlin, NH); // agg accumulator
        k_zero<<<(N + 255) / 256, 256, 0, stream>>>(deg, N);
        k_zero<<<1, 256, 0, stream>>>(stats, 2 * HID);

        long long tot = (long long)E * HID;
        k_scatter<<<(int)((tot + 255) / 256), 256, 0, stream>>>(h, src, dst, hlin, tot);
        k_deg<<<(E + 255) / 256, 256, 0, stream>>>(dst, deg, E);
        k_agg_finalize<<<(NH + 255) / 256, 256, 0, stream>>>(hlin, deg, aggb, NH);

        k_gemm<<<gemm_blocks, 256, 0, stream>>>(aggb, hb,
                                                conv_wl + (size_t)L * HID * HID,
                                                conv_bl + (size_t)L * HID,
                                                conv_wr + (size_t)L * HID * HID,
                                                hlin, N);

        k_bnstats<<<512, HID, 0, stream>>>(hlin, stats, N);
        k_bnapply<<<(NH + 255) / 256, 256, 0, stream>>>(hlin, stats,
                                                        gamma + (size_t)L * HID,
                                                        beta  + (size_t)L * HID,
                                                        h, hb, NH, 1.0f / (float)N);
    }

    k_head<<<G, 128, 0, stream>>>(h, cidx, W1, b1, W2, b2, out);
}
